// DownSampleBlock_1606317769118
// MI455X (gfx1250) — compile-verified, matrix operand order checked by reading, not run on hardware
//
#include <hip/hip_runtime.h>

typedef __attribute__((ext_vector_type(16))) _Float16 v16h;
typedef __attribute__((ext_vector_type(8)))  _Float16 v8h;
typedef __attribute__((ext_vector_type(8)))  float    v8f;

#define B_      2
#define N_      16384
#define M_      4096
#define CIN_    64
#define COUT_   128
#define NSPLIT_ 8
#define EPS_    1e-10f

// ---------------- Kernel 1: f = relu(features^T @ W1 + b1), fp32 + fp16 copies
__global__ __launch_bounds__(128) void prep_f_kernel(
    const float* __restrict__ features, const float* __restrict__ W1,
    const float* __restrict__ b1, float* __restrict__ fF32,
    _Float16* __restrict__ fH) {
  const int n = blockIdx.x, batch = blockIdx.y, c = threadIdx.x;
  __shared__ float fc[CIN_];
  if (threadIdx.x < CIN_)
    fc[threadIdx.x] = features[((size_t)batch * CIN_ + threadIdx.x) * N_ + n];
  __syncthreads();
  float acc = b1[c];
#pragma unroll
  for (int k = 0; k < CIN_; ++k) acc = fmaf(fc[k], W1[k * COUT_ + c], acc);
  acc = fmaxf(acc, 0.0f);
  const size_t o = ((size_t)batch * N_ + n) * COUT_ + c;
  fF32[o] = acc;
  fH[o]   = (_Float16)acc;
}

// ---------------- Kernel 2: W2 fp32 -> fp16
__global__ __launch_bounds__(256) void conv_w2_kernel(
    const float* __restrict__ W2, _Float16* __restrict__ W2h) {
  const int i = blockIdx.x * 256 + threadIdx.x;
  if (i < COUT_ * M_) W2h[i] = (_Float16)W2[i];
}

// Branchless online logsumexp + argmax merge (single v_exp, cndmask select).
__device__ __forceinline__ void online_update(float z, int rowIdx, float& rMax,
                                              float& rSum, int& rIdx) {
  const bool  gt = z > rMax;
  const float nm = gt ? z : rMax;
  const float sm = gt ? rMax : z;
  const float e  = __expf(sm - nm);
  rSum = gt ? __builtin_fmaf(rSum, e, 1.0f) : (rSum + e);
  rIdx = gt ? rowIdx : rIdx;
  rMax = nm;
}

// ---------------- Kernel 3: fused GEMM2 + gumbel + online softmax/argmax
// One wave per (32-wide m-tile, batch, N-split). Two 16x16 C tiles share each
// A fragment -> 8 v_wmma per n-tile, 128B-contiguous u loads per row.
__global__ __launch_bounds__(32) void gemm_gumbel_kernel(
    const _Float16* __restrict__ fH, const _Float16* __restrict__ W2h,
    const float* __restrict__ b2, const float* __restrict__ u,
    float* __restrict__ pmax, float* __restrict__ psum,
    int* __restrict__ pidx) {
  const int lane = threadIdx.x;
  const int col  = lane & 15;   // C-layout column / A-layout row
  const int kh   = lane >> 4;   // lane-group selector for A/B K interleave
  const int m0    = blockIdx.x * 32;
  const int batch = blockIdx.y;
  const int split = blockIdx.z;
  const int nPer  = N_ / NSPLIT_;
  const int nBeg  = split * nPer;

  // B fragments: W2 tile (128 x 32) resident in VGPRs for the whole n loop.
  // 16-bit B layout: halves 0..7 -> K = 8*kh + h ; halves 8..15 -> K = 16+8*kh+(h-8)
  v16h bfrag[2][4];
#pragma unroll
  for (int half = 0; half < 2; ++half) {
#pragma unroll
    for (int ks = 0; ks < 4; ++ks) {
#pragma unroll
      for (int h = 0; h < 16; ++h) {
        const int K = ks * 32 + ((h < 8) ? (8 * kh + h) : (8 + 8 * kh + h));
        bfrag[half][ks][h] = W2h[(size_t)K * M_ + m0 + 16 * half + col];
      }
    }
  }
  const float b2v0 = b2[m0 + col];
  const float b2v1 = b2[m0 + 16 + col];

  float rMax0 = -3.0e38f, rSum0 = 0.0f;
  float rMax1 = -3.0e38f, rSum1 = 0.0f;
  int   rIdx0 = 0, rIdx1 = 0;

#pragma unroll 2
  for (int nt = 0; nt < nPer; nt += 16) {
    const int n0 = nBeg + nt;
    if (nt + 16 < nPer) {  // prefetch next u tile (one row per lane)
      __builtin_prefetch(u + ((size_t)batch * N_ + n0 + 16 + col) * M_ + m0, 0, 1);
    }
    // A fragments: 16-bit A layout, two contiguous 16B runs per lane per K-step.
    const _Float16* fRow = fH + ((size_t)batch * N_ + n0 + col) * COUT_;
    v8f acc0 = {};
    v8f acc1 = {};
#pragma unroll
    for (int ks = 0; ks < 4; ++ks) {
      v8h lo = *(const v8h*)(fRow + ks * 32 + 8 * kh);
      v8h hi = *(const v8h*)(fRow + ks * 32 + 16 + 8 * kh);
      v16h a;
#pragma unroll
      for (int i = 0; i < 8; ++i) { a[i] = lo[i]; a[i + 8] = hi[i]; }
      acc0 = __builtin_amdgcn_wmma_f32_16x16x32_f16(
          false, a, false, bfrag[0][ks], (short)0, acc0, false, false);
      acc1 = __builtin_amdgcn_wmma_f32_16x16x32_f16(
          false, a, false, bfrag[1][ks], (short)0, acc1, false, false);
    }
    // z = acc + b2 + gumbel(u); branchless online (max, sum-exp, argmax).
#pragma unroll
    for (int j = 0; j < 8; ++j) {
      const int row = j + 8 * kh;  // C layout: vgpr j -> row j (lanes<16) / j+8
      const float* uRow = u + ((size_t)batch * N_ + n0 + row) * M_ + m0;
      const float u0 = __builtin_nontemporal_load(uRow + col);
      const float u1 = __builtin_nontemporal_load(uRow + 16 + col);
      const float g0 = -__logf(-__logf(u0 + EPS_) + EPS_);
      const float g1 = -__logf(-__logf(u1 + EPS_) + EPS_);
      online_update(acc0[j] + b2v0 + g0, n0 + row, rMax0, rSum0, rIdx0);
      online_update(acc1[j] + b2v1 + g1, n0 + row, rMax1, rSum1, rIdx1);
    }
  }

  // Combine lane <-> lane^16 (same m column, complementary row groups).
#pragma unroll
  for (int half = 0; half < 2; ++half) {
    float rMax = half ? rMax1 : rMax0;
    float rSum = half ? rSum1 : rSum0;
    int   rIdx = half ? rIdx1 : rIdx0;
    const float oMax = __shfl_xor(rMax, 16);
    const float oSum = __shfl_xor(rSum, 16);
    const int   oIdx = __shfl_xor(rIdx, 16);
    const float nm = fmaxf(rMax, oMax);
    const float s  = rSum * __expf(rMax - nm) + oSum * __expf(oMax - nm);
    const int   id = (oMax > rMax) ? oIdx : rIdx;
    if (lane < 16) {
      const size_t p =
          ((size_t)batch * M_ + m0 + 16 * half + col) * NSPLIT_ + split;
      pmax[p] = nm;
      psum[p] = s;
      pidx[p] = id;
    }
  }
}

// ---------------- Kernel 4: merge split partials, gather + scale outputs
__global__ __launch_bounds__(128) void finalize_kernel(
    const float* __restrict__ xyzs, const float* __restrict__ fF32,
    const float* __restrict__ pmax, const float* __restrict__ psum,
    const int* __restrict__ pidx, float* __restrict__ out) {
  const int m = blockIdx.x;
  const int batch = blockIdx.y;
  const int c = threadIdx.x;
  const size_t pb = ((size_t)batch * M_ + m) * NSPLIT_;
  float bm = -3.0e38f, bs = 0.0f;
  int bi = 0;
#pragma unroll
  for (int s2 = 0; s2 < NSPLIT_; ++s2) {
    const float pm = pmax[pb + s2];
    const float ps = psum[pb + s2];
    const int   pi = pidx[pb + s2];
    const float nm = fmaxf(bm, pm);
    bs = bs * __expf(bm - nm) + ps * __expf(pm - nm);
    bi = (pm > bm) ? pi : bi;
    bm = nm;
  }
  const float soft = 1.0f / bs;            // y_soft at the argmax
  const float v = (1.0f - soft) + soft;    // straight-through value, IEEE order

  float* xyz_out  = out;                                  // (B, M, 3)
  float* feat_out = out + (size_t)B_ * M_ * 3;            // (B, COUT, M)
  float* idx_out  = feat_out + (size_t)B_ * COUT_ * M_;   // (B, M)

  feat_out[((size_t)batch * COUT_ + c) * M_ + m] =
      v * fF32[((size_t)batch * N_ + bi) * COUT_ + c];
  if (c < 3)
    xyz_out[((size_t)batch * M_ + m) * 3 + c] =
        v * xyzs[((size_t)batch * N_ + bi) * 3 + c];
  if (c == 0) idx_out[(size_t)batch * M_ + m] = (float)bi;
}

extern "C" void kernel_launch(void* const* d_in, const int* in_sizes, int n_in,
                              void* d_out, int out_size, void* d_ws, size_t ws_size,
                              hipStream_t stream) {
  (void)in_sizes; (void)n_in; (void)out_size; (void)ws_size;
  const float* xyzs     = (const float*)d_in[0];
  const float* features = (const float*)d_in[1];
  const float* u        = (const float*)d_in[2];
  const float* W1       = (const float*)d_in[3];
  const float* b1       = (const float*)d_in[4];
  const float* W2       = (const float*)d_in[5];
  const float* b2       = (const float*)d_in[6];
  float* out = (float*)d_out;

  char* w = (char*)d_ws;
  size_t off = 0;
  float*    fF32 = (float*)(w + off);    off += (size_t)B_ * N_ * COUT_ * 4;
  _Float16* fH   = (_Float16*)(w + off); off += (size_t)B_ * N_ * COUT_ * 2;
  _Float16* W2h  = (_Float16*)(w + off); off += (size_t)COUT_ * M_ * 2;
  float*    pmax = (float*)(w + off);    off += (size_t)B_ * M_ * NSPLIT_ * 4;
  float*    psum = (float*)(w + off);    off += (size_t)B_ * M_ * NSPLIT_ * 4;
  int*      pidx = (int*)(w + off);

  prep_f_kernel<<<dim3(N_, B_), 128, 0, stream>>>(features, W1, b1, fF32, fH);
  conv_w2_kernel<<<(COUT_ * M_ + 255) / 256, 256, 0, stream>>>(W2, W2h);
  gemm_gumbel_kernel<<<dim3(M_ / 32, B_, NSPLIT_), 32, 0, stream>>>(
      fH, W2h, b2, u, pmax, psum, pidx);
  finalize_kernel<<<dim3(M_, B_), 128, 0, stream>>>(xyzs, fF32, pmax, psum, pidx, out);
}
